// MultiHeadAttention_38817914421281
// MI455X (gfx1250) — compile-verified, matrix operand order checked by reading, not run on hardware
//
#include <hip/hip_runtime.h>

// ---- CDNA5 (gfx1250, wave32) WMMA types ----
typedef __bf16 bf16_t;
typedef __attribute__((ext_vector_type(16))) __bf16 v16bf;
typedef __attribute__((ext_vector_type(8)))  float  v8f;
typedef unsigned int uint32;

// Problem dims
constexpr int Bb = 4, Tt = 2048, Cc = 1024, Hh = 16, Dd = 64;
constexpr int Mrows = Bb * Tt;       // 8192
constexpr int N3    = 3 * Cc;        // 3072

static __device__ __forceinline__ v8f wmma_bf16(v16bf a, v16bf b, v8f c) {
  return __builtin_amdgcn_wmma_f32_16x16x32_bf16(
      /*neg_a=*/false, a, /*neg_b=*/false, b,
      /*c_mod=*/(short)0, c, /*reuse_a=*/false, /*reuse_b=*/false);
}

static __device__ __forceinline__ v16bf ldv(const bf16_t* __restrict__ p, uint32 eoff) {
  return *(const v16bf*)(p + eoff);
}

// 16-lane (half-wave) row reductions: xor masks 1..8 never cross the lane-16
// boundary, matching the WMMA C/D layout (rows 0-7 in lanes 0-15, 8-15 above).
static __device__ __forceinline__ float rowmax16(float v) {
  v = fmaxf(v, __shfl_xor(v, 1));
  v = fmaxf(v, __shfl_xor(v, 2));
  v = fmaxf(v, __shfl_xor(v, 4));
  v = fmaxf(v, __shfl_xor(v, 8));
  return v;
}
static __device__ __forceinline__ float rowsum16(float v) {
  v += __shfl_xor(v, 1);
  v += __shfl_xor(v, 2);
  v += __shfl_xor(v, 4);
  v += __shfl_xor(v, 8);
  return v;
}

// ---------------- core: 32x64 per-wave tile, K=1024, double-buffered ----------
// A rows contiguous in K (stride Cc); B rows (= output columns) contiguous in K.
// unroll 2: the double-buffer rotation becomes a pure register rename between
// the two unrolled bodies (no v_mov copies, no WMMA WAR-hazard v_nops).
static __device__ __forceinline__ void gemm_tile_32x64(
    const bf16_t* __restrict__ A, const bf16_t* __restrict__ Bw,
    uint32 ao0, uint32 ao1, uint32 bo0, uint32 bo1, uint32 bo2, uint32 bo3,
    v8f acc[2][4]) {
  v16bf a0 = ldv(A, ao0),      a1 = ldv(A, ao1);
  v16bf b0 = ldv(Bw, bo0),     b1 = ldv(Bw, bo1);
  v16bf b2 = ldv(Bw, bo2),     b3 = ldv(Bw, bo3);
#pragma unroll 2
  for (uint32 k = 32; k < (uint32)Cc; k += 32) {
    // prefetch next k-step's fragments (loads return in order, so the WMMAs
    // below only need the *previous* set -> burst overlaps the new loads)
    v16bf a0n = ldv(A, ao0 + k), a1n = ldv(A, ao1 + k);
    v16bf b0n = ldv(Bw, bo0 + k), b1n = ldv(Bw, bo1 + k);
    v16bf b2n = ldv(Bw, bo2 + k), b3n = ldv(Bw, bo3 + k);
    acc[0][0] = wmma_bf16(a0, b0, acc[0][0]);
    acc[1][0] = wmma_bf16(a1, b0, acc[1][0]);
    acc[0][1] = wmma_bf16(a0, b1, acc[0][1]);
    acc[1][1] = wmma_bf16(a1, b1, acc[1][1]);
    acc[0][2] = wmma_bf16(a0, b2, acc[0][2]);
    acc[1][2] = wmma_bf16(a1, b2, acc[1][2]);
    acc[0][3] = wmma_bf16(a0, b3, acc[0][3]);
    acc[1][3] = wmma_bf16(a1, b3, acc[1][3]);
    a0 = a0n; a1 = a1n; b0 = b0n; b1 = b1n; b2 = b2n; b3 = b3n;
  }
  acc[0][0] = wmma_bf16(a0, b0, acc[0][0]);
  acc[1][0] = wmma_bf16(a1, b0, acc[1][0]);
  acc[0][1] = wmma_bf16(a0, b1, acc[0][1]);
  acc[1][1] = wmma_bf16(a1, b1, acc[1][1]);
  acc[0][2] = wmma_bf16(a0, b2, acc[0][2]);
  acc[1][2] = wmma_bf16(a1, b2, acc[1][2]);
  acc[0][3] = wmma_bf16(a0, b3, acc[0][3]);
  acc[1][3] = wmma_bf16(a1, b3, acc[1][3]);
}

// ---------------- Kernel 1: fp32 -> bf16 ----------------
__global__ void cvt_f32_bf16(const float* __restrict__ src,
                             bf16_t* __restrict__ dst, int n) {
  int i = blockIdx.x * blockDim.x + threadIdx.x;
  int stride = gridDim.x * blockDim.x;
  for (; i < n; i += stride) dst[i] = (bf16_t)src[i];
}

// ---------------- Kernel 2: QKV projection GEMM ----------------
// Block = 8 waves arranged 4(M) x 2(N): 128x128 block tile, 32x64 per wave.
// Epilogue adds bias and scatters to Q[bh,T,D], K[bh,T,D], V^T[bh,D,T].
__global__ __launch_bounds__(256) void qkv_gemm(
    const bf16_t* __restrict__ xb, const bf16_t* __restrict__ wq,
    const float* __restrict__ qkv_b, bf16_t* __restrict__ Qb,
    bf16_t* __restrict__ Kb, bf16_t* __restrict__ VTb) {
  const int lane = threadIdx.x & 31;
  const int wave = threadIdx.x >> 5;
  const int hi   = lane >> 4;
  const int ln   = lane & 15;
  const int koff = hi * 16;
  const int mbase = blockIdx.x * 128 + (wave & 3) * 32;
  const int nbase = blockIdx.y * 128 + (wave >> 2) * 64;

  const v8f zero = {0.f, 0.f, 0.f, 0.f, 0.f, 0.f, 0.f, 0.f};
  v8f acc[2][4] = {{zero, zero, zero, zero}, {zero, zero, zero, zero}};

  const uint32 ao0 = (uint32)(mbase + ln) * Cc + koff;
  const uint32 ao1 = ao0 + 16u * Cc;
  const uint32 bo0 = (uint32)(nbase + 0 * 16 + ln) * Cc + koff;
  const uint32 bo1 = (uint32)(nbase + 1 * 16 + ln) * Cc + koff;
  const uint32 bo2 = (uint32)(nbase + 2 * 16 + ln) * Cc + koff;
  const uint32 bo3 = (uint32)(nbase + 3 * 16 + ln) * Cc + koff;
  gemm_tile_32x64(xb, wq, ao0, ao1, bo0, bo1, bo2, bo3, acc);

#pragma unroll
  for (int nf = 0; nf < 4; ++nf) {
    int col   = nbase + nf * 16 + ln;       // 0..3071
    float bias = qkv_b[col];
    int which = col >> 10;                  // 0=q 1=k 2=v
    int cc    = col & 1023;
    int h = cc >> 6, d = cc & 63;
#pragma unroll
    for (int mm = 0; mm < 2; ++mm) {
#pragma unroll
      for (int r = 0; r < 8; ++r) {
        int row = mbase + mm * 16 + r + hi * 8;   // 0..8191
        int b_ = row >> 11, t = row & (Tt - 1);
        int bh = b_ * Hh + h;
        bf16_t bv = (bf16_t)(acc[mm][nf][r] + bias);
        if (which == 0)      Qb[((uint32)(bh * Tt + t)) * Dd + d] = bv;
        else if (which == 1) Kb[((uint32)(bh * Tt + t)) * Dd + d] = bv;
        else                 VTb[((uint32)(bh * Dd + d)) * Tt + t] = bv;
      }
    }
  }
}

// ---------------- Kernel 3: causal flash attention ----------------
// One wave per (bh, 16-query tile). Online softmax, KV in 32-key steps.
__global__ __launch_bounds__(32) void flash_attn(
    const bf16_t* __restrict__ Qb, const bf16_t* __restrict__ Kb,
    const bf16_t* __restrict__ VTb, bf16_t* __restrict__ AOb) {
  __shared__ __align__(32) bf16_t Pt[16 * 32];  // P tile, C-layout -> A-layout

  const int lane = threadIdx.x & 31;
  const int hi   = lane >> 4;
  const int ln   = lane & 15;
  const int koff = hi * 16;
  const int q16  = blockIdx.x * 16;
  const int bh   = blockIdx.y;
  const int b_   = bh / Hh, h = bh % Hh;

  const bf16_t* qp = Qb  + (uint32)bh * Tt * Dd;
  const bf16_t* kp = Kb  + (uint32)bh * Tt * Dd;
  const bf16_t* vp = VTb + (uint32)bh * Dd * Tt;

  // Q fragments over d=[0,32) and [32,64)
  v16bf qa0 = ldv(qp, (uint32)(q16 + ln) * Dd + koff);
  v16bf qa1 = ldv(qp, (uint32)(q16 + ln) * Dd + 32 + koff);

  const v8f zero = {0.f, 0.f, 0.f, 0.f, 0.f, 0.f, 0.f, 0.f};
  v8f acc[4] = {zero, zero, zero, zero};
  float m_i[8], l_i[8];
#pragma unroll
  for (int r = 0; r < 8; ++r) { m_i[r] = -1e30f; l_i[r] = 0.f; }

  const float scale = 0.125f;  // 1/sqrt(64)
  const int smax = (q16 + 15) >> 5;

  for (int s = 0; s <= smax; ++s) {
    const int j0 = s * 32, j1 = j0 + 16;
    if (s < smax) {  // prefetch next K/V tiles
      __builtin_prefetch(kp + (uint32)(j0 + 32 + ln) * Dd, 0, 1);
      __builtin_prefetch(vp + (uint32)ln * Tt + j0 + 32, 0, 1);
    }
    // S = Q K^T for two 16-key halves (K-dim = D = 64 -> 2 WMMAs each)
    v16bf kb0a = ldv(kp, (uint32)(j0 + ln) * Dd + koff);
    v16bf kb0b = ldv(kp, (uint32)(j0 + ln) * Dd + 32 + koff);
    v16bf kb1a = ldv(kp, (uint32)(j1 + ln) * Dd + koff);
    v16bf kb1b = ldv(kp, (uint32)(j1 + ln) * Dd + 32 + koff);
    v8f s0 = wmma_bf16(qa0, kb0a, zero);
    s0 = wmma_bf16(qa1, kb0b, s0);
    v8f s1 = wmma_bf16(qa0, kb1a, zero);
    s1 = wmma_bf16(qa1, kb1b, s1);

    const int c0 = j0 + ln, c1 = j1 + ln;
#pragma unroll
    for (int r = 0; r < 8; ++r) {
      const int rl   = r + hi * 8;     // local row in tile
      const int rowg = q16 + rl;       // global query index
      float sv0 = s0[r] * scale;
      float sv1 = s1[r] * scale;
      if (c0 > rowg) sv0 = -1e30f;     // causal mask
      if (c1 > rowg) sv1 = -1e30f;
      float mx   = rowmax16(fmaxf(sv0, sv1));
      float mnew = fmaxf(m_i[r], mx);
      float sc   = __expf(m_i[r] - mnew);
      float p0   = __expf(sv0 - mnew);
      float p1   = __expf(sv1 - mnew);
      l_i[r] = l_i[r] * sc + rowsum16(p0 + p1);
      m_i[r] = mnew;
#pragma unroll
      for (int nf = 0; nf < 4; ++nf) acc[nf][r] *= sc;
      Pt[rl * 32 + ln]      = (bf16_t)p0;
      Pt[rl * 32 + 16 + ln] = (bf16_t)p1;
    }
    asm volatile("s_wait_dscnt 0" ::: "memory");  // P stores visible in-wave

    // O += P V : A = P (16x32 keys) from LDS, B = V^T rows (contiguous keys)
    v16bf pa = *(const v16bf*)(&Pt[ln * 32 + koff]);
#pragma unroll
    for (int nf = 0; nf < 4; ++nf) {
      v16bf vb = ldv(vp, (uint32)(nf * 16 + ln) * Tt + j0 + koff);
      acc[nf] = wmma_bf16(pa, vb, acc[nf]);
    }
  }

  // Normalize and store to attn_out [B, T, C] (bf16)
#pragma unroll
  for (int nf = 0; nf < 4; ++nf) {
#pragma unroll
    for (int r = 0; r < 8; ++r) {
      int t = q16 + r + hi * 8;
      float o = acc[nf][r] / l_i[r];
      AOb[(uint32)(b_ * Tt + t) * Cc + h * Dd + nf * 16 + ln] = (bf16_t)o;
    }
  }
}

// ---------------- Kernel 4: output projection ----------------
__global__ __launch_bounds__(256) void out_gemm(
    const bf16_t* __restrict__ ab, const bf16_t* __restrict__ wo,
    const float* __restrict__ out_b, float* __restrict__ out) {
  const int lane = threadIdx.x & 31;
  const int wave = threadIdx.x >> 5;
  const int hi   = lane >> 4;
  const int ln   = lane & 15;
  const int koff = hi * 16;
  const int mbase = blockIdx.x * 128 + (wave & 3) * 32;
  const int nbase = blockIdx.y * 128 + (wave >> 2) * 64;

  const v8f zero = {0.f, 0.f, 0.f, 0.f, 0.f, 0.f, 0.f, 0.f};
  v8f acc[2][4] = {{zero, zero, zero, zero}, {zero, zero, zero, zero}};

  const uint32 ao0 = (uint32)(mbase + ln) * Cc + koff;
  const uint32 ao1 = ao0 + 16u * Cc;
  const uint32 bo0 = (uint32)(nbase + 0 * 16 + ln) * Cc + koff;
  const uint32 bo1 = (uint32)(nbase + 1 * 16 + ln) * Cc + koff;
  const uint32 bo2 = (uint32)(nbase + 2 * 16 + ln) * Cc + koff;
  const uint32 bo3 = (uint32)(nbase + 3 * 16 + ln) * Cc + koff;
  gemm_tile_32x64(ab, wo, ao0, ao1, bo0, bo1, bo2, bo3, acc);

#pragma unroll
  for (int nf = 0; nf < 4; ++nf) {
    int col = nbase + nf * 16 + ln;
    float bias = out_b[col];
#pragma unroll
    for (int mm = 0; mm < 2; ++mm) {
#pragma unroll
      for (int r = 0; r < 8; ++r) {
        int row = mbase + mm * 16 + r + hi * 8;
        out[(uint32)row * Cc + col] = acc[mm][nf][r] + bias;
      }
    }
  }
}

// ---------------- launcher ----------------
extern "C" void kernel_launch(void* const* d_in, const int* in_sizes, int n_in,
                              void* d_out, int out_size, void* d_ws, size_t ws_size,
                              hipStream_t stream) {
  const float* x     = (const float*)d_in[0];
  const float* qkv_w = (const float*)d_in[1];
  const float* qkv_b = (const float*)d_in[2];
  const float* out_w = (const float*)d_in[3];
  const float* out_b = (const float*)d_in[4];
  float* out = (float*)d_out;

  char* ws = (char*)d_ws;
  size_t off = 0;
  bf16_t* xb  = (bf16_t*)(ws + off); off += (size_t)Mrows * Cc * 2;      // 16 MB
  bf16_t* wq  = (bf16_t*)(ws + off); off += (size_t)N3 * Cc * 2;         //  6 MB
  bf16_t* wo  = (bf16_t*)(ws + off); off += (size_t)Cc * Cc * 2;         //  2 MB
  bf16_t* Qb  = (bf16_t*)(ws + off); off += (size_t)Bb * Hh * Tt * Dd * 2;
  bf16_t* Kb  = (bf16_t*)(ws + off); off += (size_t)Bb * Hh * Tt * Dd * 2;
  bf16_t* VTb = (bf16_t*)(ws + off); off += (size_t)Bb * Hh * Tt * Dd * 2;
  bf16_t* AOb = (bf16_t*)(ws + off); off += (size_t)Mrows * Cc * 2;

  cvt_f32_bf16<<<1024, 256, 0, stream>>>(x, xb, Mrows * Cc);
  cvt_f32_bf16<<<1024, 256, 0, stream>>>(qkv_w, wq, N3 * Cc);
  cvt_f32_bf16<<<512, 256, 0, stream>>>(out_w, wo, Cc * Cc);

  qkv_gemm<<<dim3(Mrows / 128, N3 / 128), 256, 0, stream>>>(xb, wq, qkv_b,
                                                            Qb, Kb, VTb);
  flash_attn<<<dim3(Tt / 16, Bb * Hh), 32, 0, stream>>>(Qb, Kb, VTb, AOb);
  out_gemm<<<dim3(Mrows / 128, Cc / 128), 256, 0, stream>>>(AOb, wo, out_b, out);
}
